// BahdanauAttn_52862457479409
// MI455X (gfx1250) — matrix operand order repaired and checked by reading, not compiled, not run on hardware
//
#include <hip/hip_runtime.h>
#include <hip/hip_bf16.h>

// Problem sizes (fixed by the reference)
#define BB 64
#define SS 4096
#define HH 128
#define DD 256               // 2H
#define NCHUNK 8             // s-chunks per batch
#define CHUNK (SS / NCHUNK)  // 512 rows per workgroup
#define WAVES 8              // 256 threads = 8 wave32
#define ROWS_PER_WAVE (CHUNK / WAVES)      // 64
#define TILES_PER_WAVE (ROWS_PER_WAVE / 16) // 4
#define WH_STRIDE 272        // 256 + 16 bf16 pad (LDS bank spread)

typedef __bf16 bf16_t;
typedef bf16_t v16bf __attribute__((ext_vector_type(16)));
typedef float  v8f   __attribute__((ext_vector_type(8)));

union V16BF { v16bf v; bf16_t e[16]; };
union V8F   { v8f   v; float  f[8];  };

__device__ __forceinline__ float fast_tanh(float x) {
    // 1 - 2/(e^{2x}+1): exact limits at +/-inf, no inf/inf NaN
    float t = __expf(2.0f * x);
    return 1.0f - 2.0f * __fdividef(1.0f, t + 1.0f);
}

__device__ __forceinline__ void cvt8(V16BF& a, int base, float4 p, float4 q) {
    a.e[base + 0] = (bf16_t)p.x; a.e[base + 1] = (bf16_t)p.y;
    a.e[base + 2] = (bf16_t)p.z; a.e[base + 3] = (bf16_t)p.w;
    a.e[base + 4] = (bf16_t)q.x; a.e[base + 5] = (bf16_t)q.y;
    a.e[base + 6] = (bf16_t)q.z; a.e[base + 7] = (bf16_t)q.w;
}

// ---------------------------------------------------------------------------
// Kernel 0: dec_feat[b,h] = concat(hidden,cell)[b] . Ws_w[h,:] + Ws_b[h]
// ---------------------------------------------------------------------------
__global__ __launch_bounds__(HH)
void dec_feat_kernel(const float* __restrict__ hidden, const float* __restrict__ cell,
                     const float* __restrict__ Ws_w, const float* __restrict__ Ws_b,
                     float* __restrict__ dec_feat) {
    const int b = blockIdx.x;
    const int h = threadIdx.x;
    const float* wr = Ws_w + h * DD;
    const float* hb = hidden + b * HH;
    const float* cb = cell + b * HH;
    float acc = Ws_b[h];
    #pragma unroll 4
    for (int d = 0; d < HH; ++d) acc += hb[d] * wr[d];
    #pragma unroll 4
    for (int d = 0; d < HH; ++d) acc += cb[d] * wr[HH + d];
    dec_feat[b * HH + h] = acc;
}

// ---------------------------------------------------------------------------
// Kernel 1: one workgroup per (chunk, batch). WMMA scores + online softmax.
// Emits: raw masked scores into attn region of d_out, per-chunk (m,l,ctx).
// ---------------------------------------------------------------------------
__global__ __launch_bounds__(256)
void attn_pass1(const float* __restrict__ enc, const unsigned char* __restrict__ mask,
                const float* __restrict__ Wh, const float* __restrict__ vw,
                const float* __restrict__ dec_feat,
                float* __restrict__ scoreOut,          // [B,S] raw masked scores
                float* __restrict__ pm, float* __restrict__ pl,
                float* __restrict__ pctx) {            // [B,NCHUNK,DD]
    extern __shared__ char lds[];
    bf16_t* whl = (bf16_t*)lds;                                   // HH * WH_STRIDE
    float* fb   = (float*)(lds + (size_t)HH * WH_STRIDE * sizeof(bf16_t));
    float* vwl      = fb;                 // HH
    float* dfl      = vwl + HH;           // HH
    float* scorebuf = dfl + HH;           // WAVES * 16 * 16
    float* ebuf     = scorebuf + WAVES * 256;  // WAVES * 16
    float* wm       = ebuf + WAVES * 16;  // WAVES
    float* wl       = wm + WAVES;         // WAVES
    float* wctx     = wl + WAVES;         // WAVES * DD

    const int b     = blockIdx.y;
    const int chunk = blockIdx.x;
    const int tid   = threadIdx.x;
    const int wave  = tid >> 5;
    const int lane  = tid & 31;
    const int half  = lane >> 4;   // 0 | 1
    const int l16   = lane & 15;
    const int koff  = half * 8;    // A-fragment K sub-offset (wave32 layout)

    // Stage Wh_w (128x256 f32) as bf16 into LDS (WMMA B operand source)
    for (int i = tid; i < HH * DD; i += 256) {
        const int r = i >> 8, c = i & (DD - 1);
        whl[r * WH_STRIDE + c] = (bf16_t)Wh[i];
    }
    if (tid < HH) { vwl[tid] = vw[tid]; dfl[tid] = dec_feat[b * HH + tid]; }
    __syncthreads();

    float m_run = -1e30f, l_run = 0.0f;
    V8F ctx;
    #pragma unroll
    for (int j = 0; j < 8; ++j) ctx.f[j] = 0.0f;

    const int waveBase = chunk * CHUNK + wave * ROWS_PER_WAVE;
    float* sb = scorebuf + wave * 256;
    float* eb = ebuf + wave * 16;
    // Per-lane base of the B operand in LDS: row h = hT*16 + l16, cols half*8.. and 16+half*8..
    const bf16_t* bbase = whl + l16 * WH_STRIDE + half * 8;
    typedef bf16_t v8bf __attribute__((ext_vector_type(8)));

    for (int t = 0; t < TILES_PER_WAVE; ++t) {
        const int tileBase = waveBase + t * 16;
        const float* rowp = enc + ((size_t)b * SS + tileBase + l16) * DD;

        // ---- enc_feat tile: k-outer, 8 independent accumulators,
        //      B fragments double-buffered so ds_load overlaps WMMA ----
        V8F acc[8];
        #pragma unroll
        for (int hT = 0; hT < 8; ++hT)
            #pragma unroll
            for (int j = 0; j < 8; ++j) acc[hT].f[j] = 0.0f;

        #pragma unroll
        for (int k = 0; k < 8; ++k) {
            // A fragment (16x32 bf16) for this K-step, converted from f32
            V16BF a;
            {
                const float4 lo  = *(const float4*)(rowp + k * 32 + koff);
                const float4 lo2 = *(const float4*)(rowp + k * 32 + koff + 4);
                const float4 hi  = *(const float4*)(rowp + k * 32 + 16 + koff);
                const float4 hi2 = *(const float4*)(rowp + k * 32 + 16 + koff + 4);
                cvt8(a, 0, lo, lo2);
                cvt8(a, 8, hi, hi2);
            }
            V16BF b0, b1;
            b0.v = __builtin_shufflevector(*(const v8bf*)(bbase + k * 32), *(const v8bf*)(bbase + k * 32 + 16), 0,1,2,3,4,5,6,7,8,9,10,11,12,13,14,15);
            #pragma unroll
            for (int hT = 0; hT < 8; ++hT) {
                if (hT < 7)
                    b1.v = __builtin_shufflevector(*(const v8bf*)(bbase + (hT + 1) * 16 * WH_STRIDE + k * 32), *(const v8bf*)(bbase + (hT + 1) * 16 * WH_STRIDE + k * 32 + 16), 0,1,2,3,4,5,6,7,8,9,10,11,12,13,14,15);
                else
                    b1 = b0;
                acc[hT].v = __builtin_amdgcn_wmma_f32_16x16x32_bf16(
                    false, a.v, false, b0.v, (short)0, acc[hT].v, false, false);
                b0 = b1;
            }
        }

        // ---- Fold enc_feat -> per-lane score partials ----
        float part[8];
        #pragma unroll
        for (int v = 0; v < 8; ++v) part[v] = 0.0f;
        #pragma unroll
        for (int hT = 0; hT < 8; ++hT) {
            const int h = hT * 16 + l16;          // D-matrix: N = lane%16
            const float vwh = vwl[h], dfh = dfl[h];
            #pragma unroll
            for (int v = 0; v < 8; ++v)           // D-matrix: M = v + 8*half
                part[v] += vwh * fast_tanh(acc[hT].f[v] + dfh);
        }

        // ---- Reduce 16 h-lanes per s-row via LDS ----
        #pragma unroll
        for (int v = 0; v < 8; ++v) sb[(half * 8 + v) * 16 + l16] = part[v];
        __syncthreads();

        float sc = -1e30f;
        if (half == 0) {
            float s = 0.0f;
            #pragma unroll
            for (int c = 0; c < 16; ++c) s += sb[l16 * 16 + c];
            const int sidx = tileBase + l16;
            if (mask[(size_t)b * SS + sidx]) s = -1e30f;
            scoreOut[(size_t)b * SS + sidx] = s;   // raw masked score
            sc = s;
        }

        // ---- Online softmax update (wave32 shuffles) ----
        float mt = sc;
        #pragma unroll
        for (int o = 16; o >= 1; o >>= 1) mt = fmaxf(mt, __shfl_xor(mt, o, 32));
        const float m_new = fmaxf(m_run, mt);
        const float corr  = __expf(m_run - m_new);
        const float e     = (half == 0) ? __expf(sc - m_new) : 0.0f;
        float lt = e;
        #pragma unroll
        for (int o = 16; o >= 1; o >>= 1) lt += __shfl_xor(lt, o, 32);
        if (half == 0) eb[l16] = e;
        m_run = m_new;
        l_run = l_run * corr + lt;
        __syncthreads();

        // ---- Context accumulation: ctx[d] = ctx[d]*corr + sum_r e_r*enc[r,d]
        // lane owns d = lane*8 .. lane*8+7 (tile just read -> WGP$/L2 hit)
        #pragma unroll
        for (int j = 0; j < 8; ++j) ctx.f[j] *= corr;
        const float* cp = enc + ((size_t)b * SS + tileBase) * DD + lane * 8;
        #pragma unroll 4
        for (int r = 0; r < 16; ++r) {
            const float w = eb[r];
            const float4 c0 = *(const float4*)(cp + (size_t)r * DD);
            const float4 c1 = *(const float4*)(cp + (size_t)r * DD + 4);
            ctx.f[0] += w * c0.x; ctx.f[1] += w * c0.y;
            ctx.f[2] += w * c0.z; ctx.f[3] += w * c0.w;
            ctx.f[4] += w * c1.x; ctx.f[5] += w * c1.y;
            ctx.f[6] += w * c1.z; ctx.f[7] += w * c1.w;
        }
    }

    // ---- Combine 8 waves -> one chunk partial (m, l, ctx[256]) ----
    if (lane == 0) { wm[wave] = m_run; wl[wave] = l_run; }
    #pragma unroll
    for (int j = 0; j < 8; ++j) wctx[wave * DD + lane * 8 + j] = ctx.f[j];
    __syncthreads();

    float M = -1e30f;
    #pragma unroll
    for (int w = 0; w < WAVES; ++w) M = fmaxf(M, wm[w]);
    float L = 0.0f, csum = 0.0f;
    #pragma unroll
    for (int w = 0; w < WAVES; ++w) {
        const float s = __expf(wm[w] - M);
        L    += wl[w] * s;
        csum += wctx[w * DD + tid] * s;
    }
    pctx[((size_t)b * NCHUNK + chunk) * DD + tid] = csum;
    if (tid == 0) {
        pm[b * NCHUNK + chunk] = M;
        pl[b * NCHUNK + chunk] = L;
    }
}

// ---------------------------------------------------------------------------
// Kernel 2: merge chunk partials -> context; normalize attn_dist in place.
// ---------------------------------------------------------------------------
__global__ __launch_bounds__(256)
void attn_finalize(const float* __restrict__ pm, const float* __restrict__ pl,
                   const float* __restrict__ pctx,
                   float* __restrict__ ctxOut, float* __restrict__ attnOut) {
    const int b = blockIdx.x;
    const int tid = threadIdx.x;

    float M = -1e30f;
    #pragma unroll
    for (int c = 0; c < NCHUNK; ++c) M = fmaxf(M, pm[b * NCHUNK + c]);
    float L = 0.0f;
    #pragma unroll
    for (int c = 0; c < NCHUNK; ++c) L += pl[b * NCHUNK + c] * __expf(pm[b * NCHUNK + c] - M);
    const float invL = __fdividef(1.0f, L);

    float csum = 0.0f;
    #pragma unroll
    for (int c = 0; c < NCHUNK; ++c)
        csum += pctx[((size_t)b * NCHUNK + c) * DD + tid] * __expf(pm[b * NCHUNK + c] - M);
    ctxOut[b * DD + tid] = csum * invL;

    for (int i = tid; i < SS; i += 256) {
        const size_t idx = (size_t)b * SS + i;
        attnOut[idx] = __expf(attnOut[idx] - M) * invL;
    }
}

// ---------------------------------------------------------------------------
extern "C" void kernel_launch(void* const* d_in, const int* in_sizes, int n_in,
                              void* d_out, int out_size, void* d_ws, size_t ws_size,
                              hipStream_t stream) {
    (void)in_sizes; (void)n_in; (void)out_size; (void)ws_size;
    const float*         enc    = (const float*)d_in[0];
    const unsigned char* mask   = (const unsigned char*)d_in[1];
    const float*         hidden = (const float*)d_in[2];
    const float*         cell   = (const float*)d_in[3];
    const float*         Wh     = (const float*)d_in[4];
    const float*         Ws     = (const float*)d_in[5];
    const float*         Wsb    = (const float*)d_in[6];
    const float*         vw     = (const float*)d_in[7];

    float* ctxOut  = (float*)d_out;           // [B, 2H]
    float* attnOut = ctxOut + BB * DD;        // [B, S]

    float* ws       = (float*)d_ws;
    float* dec_feat = ws;                     // B*H
    float* pm       = dec_feat + BB * HH;     // B*NCHUNK
    float* pl       = pm + BB * NCHUNK;       // B*NCHUNK
    float* pctx     = pl + BB * NCHUNK;       // B*NCHUNK*DD

    const size_t ldsBytes = (size_t)HH * WH_STRIDE * sizeof(bf16_t)
        + sizeof(float) * (HH + HH + WAVES * 256 + WAVES * 16 + WAVES + WAVES + WAVES * DD);

    dec_feat_kernel<<<dim3(BB), dim3(HH), 0, stream>>>(hidden, cell, Ws, Wsb, dec_feat);
    attn_pass1<<<dim3(NCHUNK, BB), dim3(256), ldsBytes, stream>>>(
        enc, mask, Wh, vw, dec_feat, attnOut, pm, pl, pctx);
    attn_finalize<<<dim3(BB), dim3(256), 0, stream>>>(pm, pl, pctx, ctxOut, attnOut);
}
